// GCNLSTMLayer_36575941492803
// MI455X (gfx1250) — matrix operand order repaired and checked by reading, not compiled, not run on hardware
//
#include <hip/hip_runtime.h>
#include <hip/hip_bf16.h>
#include <math.h>

#define NNODE 50000
#define NEDGE 800000
#define DDIM  256
#define NGATE 1024   // 4*D

typedef _Float16 v8h  __attribute__((ext_vector_type(8)));
typedef _Float16 v16h __attribute__((ext_vector_type(16)));
typedef _Float16 h4   __attribute__((ext_vector_type(4)));
typedef float    v8f  __attribute__((ext_vector_type(8)));

// ---------------------------------------------------------------------------
// Stage 1: in-degree counts for both graphs
// ---------------------------------------------------------------------------
__global__ void deg_kernel(const int* __restrict__ dst, const int* __restrict__ dst_rev,
                           float* __restrict__ deg) {
    int e = blockIdx.x * 256 + threadIdx.x;
    if (e < NEDGE) {
        atomicAdd(&deg[dst[e]], 1.0f);
        atomicAdd(&deg[NNODE + dst_rev[e]], 1.0f);
    }
}

// ---------------------------------------------------------------------------
// Stage 2: scatter-add feat[src] into agg[dst]. 64 threads/edge, float4 each.
// Edge index is wave-uniform -> readfirstlane makes src/dst scalar loads and
// the gather/atomic bases SGPRs (saddr+voffset). Accumulators (51 MB/graph)
// are L2-resident; no-return f32 atomics ride the STOREcnt path.
// ---------------------------------------------------------------------------
__global__ void scatter_kernel(const float* __restrict__ feat, const int* __restrict__ src,
                               const int* __restrict__ dst, float* __restrict__ agg) {
    int t = blockIdx.x * 256 + threadIdx.x;          // NEDGE*64 total, exact grid
    int e = __builtin_amdgcn_readfirstlane(t >> 6);  // uniform across the wave
    int c = (t & 63) << 2;
    int s = src[e], d = dst[e];                      // scalar loads
    float4 v = *(const float4*)&feat[(size_t)s * DDIM + c];
    float* p = &agg[(size_t)d * DDIM + c];
    atomicAdd(p + 0, v.x);
    atomicAdd(p + 1, v.y);
    atomicAdd(p + 2, v.z);
    atomicAdd(p + 3, v.w);
}

// ---------------------------------------------------------------------------
// Stage 3: divide by degree, build layer-0 GEMM input in f16:
//   x16[i] = [ h_in(256) | h_out(256) | feature(256) | h0[0](256) ]   K = 1024
// ---------------------------------------------------------------------------
__global__ void finalize_kernel(const float* __restrict__ agg, const float* __restrict__ deg,
                                const float* __restrict__ feat, const float* __restrict__ h0,
                                _Float16* __restrict__ x16) {
    int t = blockIdx.x * 256 + threadIdx.x;      // N*64 total
    int i = t >> 6;
    int c = (t & 63) << 2;
    if (i >= NNODE) return;
    float din  = fmaxf(deg[i], 1.0f);
    float dout = fmaxf(deg[NNODE + i], 1.0f);
    float4 a = *(const float4*)&agg[(size_t)i * DDIM + c];
    float4 b = *(const float4*)&agg[(size_t)(NNODE + i) * DDIM + c];
    float4 f = *(const float4*)&feat[(size_t)i * DDIM + c];
    float4 h = *(const float4*)&h0[(size_t)i * DDIM + c];
    _Float16* row = &x16[(size_t)i * NGATE];
    h4 oa = { (_Float16)(a.x / din),  (_Float16)(a.y / din),
              (_Float16)(a.z / din),  (_Float16)(a.w / din) };
    h4 ob = { (_Float16)(b.x / dout), (_Float16)(b.y / dout),
              (_Float16)(b.z / dout), (_Float16)(b.w / dout) };
    h4 of = { (_Float16)f.x, (_Float16)f.y, (_Float16)f.z, (_Float16)f.w };
    h4 oh = { (_Float16)h.x, (_Float16)h.y, (_Float16)h.z, (_Float16)h.w };
    *(h4*)&row[c]        = oa;
    *(h4*)&row[256 + c]  = ob;
    *(h4*)&row[512 + c]  = of;
    *(h4*)&row[768 + c]  = oh;
}

// ---------------------------------------------------------------------------
// Stage 4: weights -> f16, concatenated along K (power-of-two K -> shift/mask).
// Row-major W is already the WMMA B-fragment-friendly layout (B = W^T), so no
// transpose anywhere.
// ---------------------------------------------------------------------------
__global__ void wcat_kernel(const float* __restrict__ Wih, const float* __restrict__ Whh,
                            _Float16* __restrict__ w16, int kshift, int in_f) {
    int idx = blockIdx.x * 256 + threadIdx.x;
    int K = 1 << kshift;
    if (idx >= NGATE << kshift) return;
    int j = idx >> kshift, k = idx & (K - 1);
    float v = (k < in_f) ? Wih[(size_t)j * in_f + k]
                         : Whh[(size_t)j * DDIM + (k - in_f)];
    w16[idx] = (_Float16)v;
}

// ---------------------------------------------------------------------------
// Stage 5: WMMA GEMM  C[M,1024] = A[M,K] * W[1024,K]^T   (f16 in, f32 acc)
// Block tile 128x128, 8 waves (4x2), wave tile 32x64 = 2x4 wmma_f32_16x16x32_f16.
// Double-buffered LDS; next K-tile is register-prefetched before the WMMA
// block so the s_wait_loadcnt lands after the compute. LDS rows padded to
// 40 halves (80 B) -> conflict-free ds_load_b128 fragment reads.
// Epilogue predication is per-16-row-tile and uniform (NNODE % 16 == 0).
// K is a template parameter -> shift-based addressing, folded loop bounds.
// ---------------------------------------------------------------------------
#define BM 128
#define BN 128
#define KT 32

template <int K>
__launch_bounds__(256)
__global__ void gemm_kernel(const _Float16* __restrict__ A, const _Float16* __restrict__ W,
                            float* __restrict__ C) {
    __shared__ __align__(16) _Float16 As[2][BM][40];
    __shared__ __align__(16) _Float16 Ws[2][BN][40];
    const int tid  = threadIdx.x;
    const int lane = tid & 31;
    const int wave = tid >> 5;
    const int wm   = wave >> 1;          // 0..3 -> 32 rows each
    const int wn   = wave & 1;           // 0..1 -> 64 cols each
    const int m0   = blockIdx.x * BM;
    const int n0   = blockIdx.y * BN;
    const int half = lane >> 4;          // lane group 0/1
    const int lr   = lane & 15;

    // staging-chunk coordinates (2 x 16B chunks per thread per tile)
    const int r0 = tid >> 2;             // 0..63
    const int r1 = r0 + 64;              // 64..127
    const int cc = (tid & 3) << 3;       // 0,8,16,24 halves
    int ga0 = m0 + r0; ga0 = (ga0 < NNODE) ? ga0 : (NNODE - 1);
    int ga1 = m0 + r1; ga1 = (ga1 < NNODE) ? ga1 : (NNODE - 1);

    v8f acc[2][4] = {};

    // preload K-tile 0
    uint4 a0 = *(const uint4*)&A[(size_t)ga0 * K + cc];
    uint4 a1 = *(const uint4*)&A[(size_t)ga1 * K + cc];
    uint4 w0 = *(const uint4*)&W[(size_t)(n0 + r0) * K + cc];
    uint4 w1 = *(const uint4*)&W[(size_t)(n0 + r1) * K + cc];
    *(uint4*)&As[0][r0][cc] = a0;
    *(uint4*)&As[0][r1][cc] = a1;
    *(uint4*)&Ws[0][r0][cc] = w0;
    *(uint4*)&Ws[0][r1][cc] = w1;
    __syncthreads();

    int cur = 0;
    for (int kb = 0; kb < K; kb += KT) {
        const int nxt = kb + KT;
        const bool has = nxt < K;
        // issue next tile's global loads; latency overlaps the WMMAs below
        if (has) {
            a0 = *(const uint4*)&A[(size_t)ga0 * K + nxt + cc];
            a1 = *(const uint4*)&A[(size_t)ga1 * K + nxt + cc];
            w0 = *(const uint4*)&W[(size_t)(n0 + r0) * K + nxt + cc];
            w1 = *(const uint4*)&W[(size_t)(n0 + r1) * K + nxt + cc];
        }

        // fragments from LDS
        v16h af[2], wf[4];
#pragma unroll
        for (int t = 0; t < 2; ++t) {
            // A frag (16x32): lane holds row lr, k = 16*(e/8) + 8*half + e%8
            const _Float16* ap = &As[cur][wm * 32 + t * 16 + lr][0];
            v8h alo = *(const v8h*)(ap + 8 * half);
            v8h ahi = *(const v8h*)(ap + 16 + 8 * half);
            af[t] = __builtin_shufflevector(alo, ahi,
                     0,1,2,3,4,5,6,7,8,9,10,11,12,13,14,15);
        }
#pragma unroll
        for (int u = 0; u < 4; ++u) {
            // B frag (32x16): lane holds col lr (= row of W), k = 16*(e/8) + 8*half + e%8
            const _Float16* wp = &Ws[cur][wn * 64 + u * 16 + lr][0];
            v8h wlo = *(const v8h*)(wp + 8 * half);
            v8h whi = *(const v8h*)(wp + 16 + 8 * half);
            wf[u] = __builtin_shufflevector(wlo, whi,
                     0,1,2,3,4,5,6,7,8,9,10,11,12,13,14,15);
        }
#pragma unroll
        for (int t = 0; t < 2; ++t)
#pragma unroll
            for (int u = 0; u < 4; ++u)
                acc[t][u] = __builtin_amdgcn_wmma_f32_16x16x32_f16(
                    false, af[t], false, wf[u], (short)0, acc[t][u], false, false);

        if (has) {
            __syncthreads();                 // everyone done reading buf[1-cur]
            *(uint4*)&As[1 - cur][r0][cc] = a0;
            *(uint4*)&As[1 - cur][r1][cc] = a1;
            *(uint4*)&Ws[1 - cur][r0][cc] = w0;
            *(uint4*)&Ws[1 - cur][r1][cc] = w1;
            __syncthreads();                 // new tile visible
            cur ^= 1;
        }
    }

    // Epilogue: C/D layout VGPR r -> row r + 8*half, col = lane%16.
    // NNODE % 16 == 0 -> whole 16-row tiles valid/invalid: uniform branch.
#pragma unroll
    for (int t = 0; t < 2; ++t) {
        const int tb = m0 + wm * 32 + t * 16;
        if (tb < NNODE) {
            const int rowb = tb + half * 8;
#pragma unroll
            for (int u = 0; u < 4; ++u) {
                const int col = n0 + wn * 64 + u * 16 + lr;
#pragma unroll
                for (int r = 0; r < 8; ++r)
                    C[(size_t)(rowb + r) * NGATE + col] = acc[t][u][r];
            }
        }
    }
}

// ---------------------------------------------------------------------------
// Stage 6: LSTM pointwise (float4-vectorized). Gate order i,f,g,o. Writes h/c
// to d_out slices; also emits next layer's f16 input [h2 | h0[l+1]] (K=512).
// ---------------------------------------------------------------------------
__device__ __forceinline__ float sig1(float x) { return 1.0f / (1.0f + __expf(-x)); }

__global__ void lstm_kernel(const float* __restrict__ gates, const float* __restrict__ b_ih,
                            const float* __restrict__ b_hh, const float* __restrict__ c0,
                            const float* __restrict__ h0_next, _Float16* __restrict__ x16_next,
                            float* __restrict__ h_out, float* __restrict__ c_out,
                            float* __restrict__ final_out) {
    int t = blockIdx.x * 256 + threadIdx.x;      // N*64 total
    int i = t >> 6;
    int j = (t & 63) << 2;
    if (i >= NNODE) return;
    const float* g = &gates[(size_t)i * NGATE];
    float4 gi = *(const float4*)&g[j];
    float4 gf = *(const float4*)&g[256 + j];
    float4 gg = *(const float4*)&g[512 + j];
    float4 go = *(const float4*)&g[768 + j];
    float4 bii = *(const float4*)&b_ih[j],       bhi = *(const float4*)&b_hh[j];
    float4 bif = *(const float4*)&b_ih[256 + j], bhf = *(const float4*)&b_hh[256 + j];
    float4 big = *(const float4*)&b_ih[512 + j], bhg = *(const float4*)&b_hh[512 + j];
    float4 bio = *(const float4*)&b_ih[768 + j], bho = *(const float4*)&b_hh[768 + j];
    float4 cv  = *(const float4*)&c0[(size_t)i * DDIM + j];

    float4 h2, c2;
    {
        float is, fs, gt, os, cc2;
#define LSTM_COMP(X) \
        is  = sig1(gi.X + bii.X + bhi.X); \
        fs  = sig1(gf.X + bif.X + bhf.X); \
        gt  = tanhf(gg.X + big.X + bhg.X); \
        os  = sig1(go.X + bio.X + bho.X); \
        cc2 = fs * cv.X + is * gt; \
        c2.X = cc2; \
        h2.X = os * tanhf(cc2);
        LSTM_COMP(x) LSTM_COMP(y) LSTM_COMP(z) LSTM_COMP(w)
#undef LSTM_COMP
    }

    *(float4*)&h_out[(size_t)i * DDIM + j] = h2;
    *(float4*)&c_out[(size_t)i * DDIM + j] = c2;
    if (x16_next) {
        float4 hn = *(const float4*)&h0_next[(size_t)i * DDIM + j];
        h4 oh  = { (_Float16)h2.x, (_Float16)h2.y, (_Float16)h2.z, (_Float16)h2.w };
        h4 ohn = { (_Float16)hn.x, (_Float16)hn.y, (_Float16)hn.z, (_Float16)hn.w };
        *(h4*)&x16_next[(size_t)i * 512 + j]       = oh;
        *(h4*)&x16_next[(size_t)i * 512 + 256 + j] = ohn;
    }
    if (final_out) *(float4*)&final_out[(size_t)i * DDIM + j] = h2;
}

// ---------------------------------------------------------------------------
extern "C" void kernel_launch(void* const* d_in, const int* in_sizes, int n_in,
                              void* d_out, int out_size, void* d_ws, size_t ws_size,
                              hipStream_t stream) {
    const float* feature = (const float*)d_in[0];
    const int*   esrc    = (const int*)d_in[1];
    const int*   edst    = (const int*)d_in[2];
    const int*   esrcr   = (const int*)d_in[3];
    const int*   edstr   = (const int*)d_in[4];
    const float* h0      = (const float*)d_in[5];
    const float* c0      = (const float*)d_in[6];
    const float* Wih[3]  = {(const float*)d_in[7],  (const float*)d_in[11], (const float*)d_in[15]};
    const float* Whh[3]  = {(const float*)d_in[8],  (const float*)d_in[12], (const float*)d_in[16]};
    const float* bih[3]  = {(const float*)d_in[9],  (const float*)d_in[13], (const float*)d_in[17]};
    const float* bhh[3]  = {(const float*)d_in[10], (const float*)d_in[14], (const float*)d_in[18]};
    float* out = (float*)d_out;                       // [output | hs(3) | cs(3)], each N*D

    const size_t ND = (size_t)NNODE * DDIM;
    char* ws = (char*)d_ws;
    const size_t G = (size_t)NNODE * NGATE * 4;       // 204.8 MB region
    float*    agg   = (float*)ws;                     // 2*N*D f32 (overlaid w/ gates)
    float*    deg   = (float*)(ws + 2 * ND * 4);      // 2*N f32
    float*    gates = (float*)ws;                     // reuse after finalize
    _Float16* x16   = (_Float16*)(ws + G);            // N*1024 f16
    _Float16* w16_0 = (_Float16*)(ws + G + (size_t)NNODE * NGATE * 2);
    _Float16* w16_1 = w16_0 + (size_t)NGATE * 1024;
    _Float16* w16_2 = w16_1 + (size_t)NGATE * 512;

    // zero aggregation accumulators + degrees
    hipMemsetAsync(ws, 0, 2 * ND * 4 + 2 * (size_t)NNODE * 4, stream);

    deg_kernel<<<(NEDGE + 255) / 256, 256, 0, stream>>>(edst, edstr, deg);
    scatter_kernel<<<(NEDGE * 64) / 256, 256, 0, stream>>>(feature, esrc,  edst,  agg);
    scatter_kernel<<<(NEDGE * 64) / 256, 256, 0, stream>>>(feature, esrcr, edstr, agg + ND);
    finalize_kernel<<<(NNODE * 64) / 256, 256, 0, stream>>>(agg, deg, feature, h0, x16);

    wcat_kernel<<<(NGATE * 1024 + 255) / 256, 256, 0, stream>>>(Wih[0], Whh[0], w16_0, 10, 768);
    wcat_kernel<<<(NGATE * 512  + 255) / 256, 256, 0, stream>>>(Wih[1], Whh[1], w16_1, 9, 256);
    wcat_kernel<<<(NGATE * 512  + 255) / 256, 256, 0, stream>>>(Wih[2], Whh[2], w16_2, 9, 256);

    dim3 ggrid((NNODE + BM - 1) / BM, NGATE / BN);   // (391, 8)
    // layer 0
    gemm_kernel<1024><<<ggrid, 256, 0, stream>>>(x16, w16_0, gates);
    lstm_kernel<<<(NNODE * 64) / 256, 256, 0, stream>>>(gates, bih[0], bhh[0], c0,
                                           h0 + ND, x16, out + ND, out + 4 * ND, nullptr);
    // layer 1
    gemm_kernel<512><<<ggrid, 256, 0, stream>>>(x16, w16_1, gates);
    lstm_kernel<<<(NNODE * 64) / 256, 256, 0, stream>>>(gates, bih[1], bhh[1], c0 + ND,
                                           h0 + 2 * ND, x16, out + 2 * ND, out + 5 * ND, nullptr);
    // layer 2
    gemm_kernel<512><<<ggrid, 256, 0, stream>>>(x16, w16_2, gates);
    lstm_kernel<<<(NNODE * 64) / 256, 256, 0, stream>>>(gates, bih[2], bhh[2], c0 + 2 * ND,
                                           nullptr, nullptr, out + 3 * ND, out + 6 * ND, out);
}
